// ESRNN_65824668779039
// MI455X (gfx1250) — matrix operand order repaired and checked by reading, not compiled, not run on hardware
//
#include <hip/hip_runtime.h>
#include <hip/hip_bf16.h>

// ---------------------------------------------------------------------------
// ES-RNN for MI455X (gfx1250): WMMA f16 16x16x32 everywhere a GEMM appears.
// B=64, S=512, H=128, NH=4 (hd=32), P=7, PRED=5.
// ---------------------------------------------------------------------------

typedef __attribute__((ext_vector_type(16))) _Float16 v16h;
typedef __attribute__((ext_vector_type(8)))  float    v8f;
typedef unsigned int u32x4 __attribute__((ext_vector_type(4)));
typedef int          i32x4 __attribute__((ext_vector_type(4)));
typedef int          i32x8 __attribute__((ext_vector_type(8)));

#if defined(__gfx1250__) && __has_builtin(__builtin_amdgcn_tensor_load_to_lds)
#define HAVE_TDM 1
#endif

#if __has_builtin(__builtin_amdgcn_sched_barrier)
#define SCHED_FENCE() __builtin_amdgcn_sched_barrier(0)
#else
#define SCHED_FENCE() asm volatile("")  // inline asm = scheduling boundary
#endif

union F16Frag { uint4 u4[2]; v16h h; };

__device__ __forceinline__ float sigf(float x) { return 1.0f / (1.0f + __expf(-x)); }

// Branch-free tanh: hardware v_tanh_f32 when exposed, else 1 - 2/(e^2x + 1)
// (one v_exp_f32 + rcp; libm tanhf emits a divergent EXEC branch that would
// sit on the LSTM recurrence's critical chain).
__device__ __forceinline__ float tanh_fast(float x) {
#if __has_builtin(__builtin_amdgcn_tanhf)
  return __builtin_amdgcn_tanhf(x);
#else
  float e = __expf(2.0f * x);
  return 1.0f - 2.0f / (e + 1.0f);
#endif
}

// --- WMMA wrapper ----------------------------------------------------------
__device__ __forceinline__ v8f wmma_f16(v16h a, v16h b, v8f c) {
  // v_wmma_f32_16x16x32_f16 : D = A(16x32 f16) * B(32x16 f16) + C(16x16 f32)
  return __builtin_amdgcn_wmma_f32_16x16x32_f16(
      /*neg_a=*/false, a, /*neg_b=*/false, b,
      /*c_mod=*/(short)0, c, /*reuse_a=*/false, /*reuse_b=*/false);
}

// --- A-operand fragment (16x32, f16), source row-major M x K (K contiguous).
__device__ __forceinline__ v16h load_a_frag(const _Float16* A, int ldk,
                                            int m0, int k0, int lane) {
  int l  = lane & 15;
  int hi = (lane >> 4) & 1;
  const uint32_t* p =
      (const uint32_t*)(A + (size_t)(m0 + l) * ldk + k0 + hi * 8);
  F16Frag f;
  f.u4[0] = *(const uint4*)(p);       // K = base .. base+7
  f.u4[1] = *(const uint4*)(p + 8);   // K = base+16 .. base+23
  return f.h;
}

// --- B-operand fragment (32x16, f16) from W stored row-major N x K.
__device__ __forceinline__ v16h load_b_frag(const _Float16* W, int ldk,
                                            int n0, int k0, int lane) {
  int l  = lane & 15;
  int hi = (lane >> 4) & 1;
  const uint32_t* p =
      (const uint32_t*)(W + (size_t)(n0 + l) * ldk + k0 + hi * 8);
  F16Frag f;
  f.u4[0] = *(const uint4*)(p);
  f.u4[1] = *(const uint4*)(p + 8);
  return f.h;
}

// C/D fragment element (r, lane): row = r + (lane>=16 ? 8 : 0), col = lane&15.

// ===========================================================================
// 1) Exponential-smoothing components: one thread per batch, sequential scan.
// ===========================================================================
__global__ void es_kernel(const float* __restrict__ x, const float* alpha,
                          const float* beta, const float* gamma,
                          float* __restrict__ es) {
  int b = threadIdx.x;
  if (b >= 64) return;
  float a  = sigf(alpha[0]);
  float bt = sigf(beta[0]);
  float g  = sigf(gamma[0]);
  const float* xb = x + (size_t)b * 512 * 4;   // series = x[:, :, 0]
  float ring[7];
  float lvl = xb[0];
  float tr  = xb[4] - xb[0];
#pragma unroll
  for (int i = 0; i < 7; ++i) ring[i] = xb[i * 4] - lvl;
  float* eb = es + (size_t)b * 512 * 3;
  eb[0] = lvl; eb[1] = tr; eb[2] = ring[0];
  for (int t = 1; t < 512; ++t) {
    int tm = t % 7;
    float xt = xb[t * 4];
    float prev = ring[tm];
    float nl = a * (xt - prev) + (1.0f - a) * (lvl + tr);
    float nt = bt * (nl - lvl) + (1.0f - bt) * tr;
    float ns = g * (xt - nl) + (1.0f - g) * prev;
    ring[tm] = ns; lvl = nl; tr = nt;
    eb[t * 3 + 0] = nl; eb[t * 3 + 1] = nt; eb[t * 3 + 2] = ns;
  }
}

// ===========================================================================
// 2) Prep: fold proj into Wih0 -> Weff (512x4); combined biases.
// ===========================================================================
__global__ void weff_kernel(const float* __restrict__ Wih0,
                            const float* __restrict__ projW,
                            const float* __restrict__ projb,
                            const float* bih0, const float* bhh0,
                            const float* bih1, const float* bhh1,
                            float* __restrict__ weff, float* __restrict__ bias0,
                            float* __restrict__ bias1) {
  int n = blockIdx.x * blockDim.x + threadIdx.x;
  if (n >= 512) return;
  float w0 = 0, w1 = 0, w2 = 0, w3 = 0, bb = 0;
  for (int k = 0; k < 128; ++k) {
    float wv = Wih0[n * 131 + k];
    w0 += wv * projW[k * 4 + 0];
    w1 += wv * projW[k * 4 + 1];
    w2 += wv * projW[k * 4 + 2];
    w3 += wv * projW[k * 4 + 3];
    bb += wv * projb[k];
  }
  weff[n * 4 + 0] = w0; weff[n * 4 + 1] = w1;
  weff[n * 4 + 2] = w2; weff[n * 4 + 3] = w3;
  bias0[n] = bih0[n] + bhh0[n] + bb;
  bias1[n] = bih1[n] + bhh1[n];
}

__global__ void f32_to_f16(const float* __restrict__ s, _Float16* __restrict__ d,
                           int n) {
  int i = blockIdx.x * blockDim.x + threadIdx.x;
  if (i < n) d[i] = (_Float16)s[i];
}

// ===========================================================================
// 3) xg0[t][b][n] (time-major): folded K=7 contraction.
// ===========================================================================
__global__ void xg0_kernel(const float* __restrict__ x,
                           const float* __restrict__ es,
                           const float* __restrict__ weff,
                           const float* __restrict__ Wih0,
                           const float* __restrict__ bias0,
                           float* __restrict__ xg) {
  size_t i = (size_t)blockIdx.x * blockDim.x + threadIdx.x;
  if (i >= (size_t)32768 * 512) return;
  int n = (int)(i & 511);
  size_t m = i >> 9;  // m = b*512 + t
  const float* xr = x + m * 4;
  const float* er = es + m * 3;
  float v = bias0[n]
          + xr[0] * weff[n * 4 + 0] + xr[1] * weff[n * 4 + 1]
          + xr[2] * weff[n * 4 + 2] + xr[3] * weff[n * 4 + 3]
          + er[0] * Wih0[n * 131 + 128] + er[1] * Wih0[n * 131 + 129]
          + er[2] * Wih0[n * 131 + 130];
  size_t t = m & 511, b = m >> 9;
  xg[(t * 64 + b) * 512 + n] = v;
}

// ===========================================================================
// 4) Persistent single-workgroup LSTM recurrence. 512 threads = 16 waves.
//    Whh staged to LDS via the Tensor Data Mover (fallback: b128 copies).
//    The k-loop is manually software-pipelined: the next k-step's four B
//    fragments are loaded before the current k-step's WMMAs, separated by a
//    hard sched_barrier so the scheduler cannot re-serialize them.
// ===========================================================================
__global__ void __launch_bounds__(512)
lstm_kernel(const float* __restrict__ xg, const _Float16* __restrict__ Whh,
            _Float16* __restrict__ hout) {
  extern __shared__ __align__(16) char smem[];
  _Float16* hbuf = (_Float16*)smem;                    // 64*128 f16 = 16KB
  _Float16* wlds = (_Float16*)(smem + 64 * 128 * 2);   // 512*128 f16 = 128KB

  int tid = threadIdx.x;
  int lane = tid & 31;
  int wave = tid >> 5;
  int mtile = wave & 3;
  int jt0 = (wave >> 2) << 1;
  int m0 = mtile << 4;
  int col = lane & 15;
  int roff = (lane >> 4) << 3;

#ifdef HAVE_TDM
  // TDM: 2D tensor 512 rows x 128 cols, 2B elements, whole-tensor tile,
  // destination LDS offset = 16KB (just past hbuf).  D# packed per ISA 8.3-8.6.
  if (wave == 0) {
    unsigned long long ga = (unsigned long long)(uintptr_t)Whh;
    u32x4 g0;
    g0.x = 1u;                                   // count=1, user mode
    g0.y = (unsigned)(64 * 128 * 2);             // lds_addr (bytes)
    g0.z = (unsigned)(ga & 0xFFFFFFFFu);         // global_addr[31:0]
    g0.w = (unsigned)((ga >> 32) & 0x1FFFFFFu)   // global_addr[56:32]
         | (2u << 30);                           // type=2 ("image")
    i32x8 g1;
    g1[0] = (int)(1u << 16);        // workgroup_mask=0, data_size=1 (2B)
    g1[1] = (int)(128u << 16);      // tensor_dim0 = 128   (bits 79:48)
    g1[2] = (int)(512u << 16);      // tensor_dim1 = 512   (bits 111:80)
    g1[3] = (int)(128u << 16);      // tile_dim0  = 128    (bits 127:112)
    g1[4] = 512;                    // tile_dim1  = 512    (bits 143:128)
    g1[5] = 128;                    // tensor_dim0_stride = 128 (bits 207:160)
    g1[6] = 0;                      // dim1_stride[15:0] (65536 -> low16 = 0)
    g1[7] = 1;                      // dim1_stride[47:16] = 1 (=> 65536)
    i32x4 gz = {0, 0, 0, 0};
#if __clang_major__ >= 23
    i32x8 gz8 = {0, 0, 0, 0, 0, 0, 0, 0};
    __builtin_amdgcn_tensor_load_to_lds(g0, g1, gz, gz, gz8, 0);
#else
    __builtin_amdgcn_tensor_load_to_lds(g0, g1, gz, gz, 0);
#endif
#if __has_builtin(__builtin_amdgcn_s_wait_tensorcnt)
    __builtin_amdgcn_s_wait_tensorcnt((short)0);
#endif
  }
#else
  for (int i = tid; i < 512 * 128 / 8; i += 512)
    ((uint4*)wlds)[i] = ((const uint4*)Whh)[i];
#endif
  for (int i = tid; i < 64 * 128 / 8; i += 512)
    ((uint4*)hbuf)[i] = uint4{0, 0, 0, 0};
  v8f c[2];
  c[0] = v8f{}; c[1] = v8f{};
  __syncthreads();

  for (int t = 0; t < 512; ++t) {
    // Treat all memory as modified each step: keeps fragment loads in-loop
    // (cheap LDS reloads) instead of hoisted-and-spilled to scratch.
    asm volatile("" ::: "memory");

    v16h af[4];
#pragma unroll
    for (int kk = 0; kk < 4; ++kk)
      af[kk] = load_a_frag(hbuf, 128, m0, kk * 32, lane);
    __syncthreads();  // reads of h_{t-1} done before anyone writes h_t

    if (t + 1 < 512) {
      // 512 threads x 256B = exactly next step's 128KB xg slab
      const char* pb = (const char*)(xg + (size_t)(t + 1) * 32768);
      __builtin_prefetch(pb + tid * 256, 0, 3);
    }

    const float* xgt = xg + (size_t)t * 32768;  // [b][n] slab for step t
#pragma unroll
    for (int task = 0; task < 2; ++task) {
      int j0 = (jt0 + task) << 4;
      v8f g4[4];
      // seed accumulators with this step's gate inputs (global, overlapped)
#pragma unroll
      for (int gt = 0; gt < 4; ++gt) {
        int n0 = gt * 128 + j0;
#pragma unroll
        for (int r = 0; r < 8; ++r)
          g4[gt][r] = xgt[(size_t)(m0 + r + roff) * 512 + n0 + col];
      }
      // software pipeline: B fragments for k-step kk+1 are loaded before
      // the WMMAs of k-step kk; the sched fence keeps it that way.
      v16h b0 = load_b_frag(wlds, 128, 0 * 128 + j0, 0, lane);
      v16h b1 = load_b_frag(wlds, 128, 1 * 128 + j0, 0, lane);
      v16h b2 = load_b_frag(wlds, 128, 2 * 128 + j0, 0, lane);
      v16h b3 = load_b_frag(wlds, 128, 3 * 128 + j0, 0, lane);
#pragma unroll
      for (int kk = 0; kk < 4; ++kk) {
        v16h n0 = b0, n1 = b1, n2 = b2, n3 = b3;
        if (kk < 3) {
          int k1 = (kk + 1) * 32;
          n0 = load_b_frag(wlds, 128, 0 * 128 + j0, k1, lane);
          n1 = load_b_frag(wlds, 128, 1 * 128 + j0, k1, lane);
          n2 = load_b_frag(wlds, 128, 2 * 128 + j0, k1, lane);
          n3 = load_b_frag(wlds, 128, 3 * 128 + j0, k1, lane);
        }
        SCHED_FENCE();  // loads above stay above; wmmas below stay below
        g4[0] = wmma_f16(af[kk], b0, g4[0]);
        g4[1] = wmma_f16(af[kk], b1, g4[1]);
        g4[2] = wmma_f16(af[kk], b2, g4[2]);
        g4[3] = wmma_f16(af[kk], b3, g4[3]);
        b0 = n0; b1 = n1; b2 = n2; b3 = n3;
      }
#pragma unroll
      for (int r = 0; r < 8; ++r) {
        float ig = sigf(g4[0][r]);
        float fg = sigf(g4[1][r]);
        float gg = tanh_fast(g4[2][r]);
        float og = sigf(g4[3][r]);
        float cn = fg * c[task][r] + ig * gg;
        c[task][r] = cn;
        float hn = og * tanh_fast(cn);
        int m = m0 + r + roff;
        hbuf[m * 128 + j0 + col] = (_Float16)hn;
        hout[((size_t)m * 512 + t) * 128 + j0 + col] = (_Float16)hn;
      }
    }
    __syncthreads();  // h_t visible before next step's reads
  }
}

// ===========================================================================
// 5) Generic WMMA GEMM, K compile-time so the k-loop fully unrolls and all
//    fragment loads batch ahead of the WMMA chain.
//    REMAP_TS: row m=b*512+t written to row t*64+b (time-major for LSTM).
// ===========================================================================
template <bool OUT_F16, bool REMAP_TS, int K>
__global__ void __launch_bounds__(256)
gemm_kernel(const _Float16* __restrict__ A, const _Float16* __restrict__ W,
            const float* __restrict__ bias, void* __restrict__ out,
            int M, int N) {
  int lane = threadIdx.x & 31;
  int wid = blockIdx.x * (blockDim.x >> 5) + (threadIdx.x >> 5);
  int ntiles = N >> 4;
  int total = (M >> 4) * ntiles;
  if (wid >= total) return;  // wave-uniform: EXEC stays all-ones
  int m0 = (wid / ntiles) << 4;
  int n0 = (wid % ntiles) << 4;
  int col = lane & 15;
  int roff = (lane >> 4) << 3;
  float bv = bias ? bias[n0 + col] : 0.0f;
  v8f acc = {bv, bv, bv, bv, bv, bv, bv, bv};
#pragma unroll
  for (int kk = 0; kk < K; kk += 32)
    acc = wmma_f16(load_a_frag(A, K, m0, kk, lane),
                   load_b_frag(W, K, n0, kk, lane), acc);
#pragma unroll
  for (int r = 0; r < 8; ++r) {
    int m = m0 + r + roff;
    size_t orow = REMAP_TS ? ((size_t)(m & 511) * 64 + (m >> 9)) : (size_t)m;
    if constexpr (OUT_F16)
      ((_Float16*)out)[orow * N + n0 + col] = (_Float16)acc[r];
    else
      ((float*)out)[orow * N + n0 + col] = acc[r];
  }
}

// ===========================================================================
// 6) Flash attention. Block = one (b, head); 256 threads = 8 waves.
// ===========================================================================
__global__ void __launch_bounds__(256)
attn_kernel(const _Float16* __restrict__ qkv, _Float16* __restrict__ attnh) {
  __shared__ __align__(16) _Float16 Vsw[32 * 512];      // V^T: [d][key]
  __shared__ __align__(16) _Float16 Pbuf[8][16 * 32];   // per-wave P tile
  int b = blockIdx.x >> 2;
  int h = blockIdx.x & 3;
  int tid = threadIdx.x;
  int lane = tid & 31;
  int wave = tid >> 5;
  const size_t base = (size_t)b * 512 * 384;

  for (int i = tid; i < 512 * 32; i += 256) {
    int key = i >> 5, d = i & 31;
    Vsw[d * 512 + key] = qkv[base + (size_t)key * 384 + 256 + h * 32 + d];
  }
  __syncthreads();

  const _Float16* Qb = qkv + base + h * 32;        // row stride 384
  const _Float16* Kb = qkv + base + 128 + h * 32;  // row stride 384
  const float scale = 0.17677669529663687f;        // 1/sqrt(32)
  int col = lane & 15;
  int roff = (lane >> 4) << 3;

  for (int qi = 0; qi < 4; ++qi) {
    int q0 = (wave + qi * 8) << 4;  // 32 query tiles over 8 waves
    v16h qf = load_a_frag(Qb, 384, q0, 0, lane);
    float mrun[8], lrun[8];
    v8f O0 = v8f{}, O1 = v8f{};
#pragma unroll
    for (int r = 0; r < 8; ++r) { mrun[r] = -1e30f; lrun[r] = 0.0f; }

    for (int kp = 0; kp < 16; ++kp) {
      int key0 = kp * 32;
      v8f z = v8f{};
      v8f s0 = wmma_f16(qf, load_b_frag(Kb, 384, key0, 0, lane), z);
      v8f s1 = wmma_f16(qf, load_b_frag(Kb, 384, key0 + 16, 0, lane), z);
      _Float16* P = &Pbuf[wave][0];
#pragma unroll
      for (int r = 0; r < 8; ++r) {
        float a0 = s0[r] * scale, a1 = s1[r] * scale;
        float mx = fmaxf(a0, a1);
        for (int off = 1; off < 16; off <<= 1)
          mx = fmaxf(mx, __shfl_xor(mx, off, 32));
        float mnew = fmaxf(mrun[r], mx);
        float corr = __expf(mrun[r] - mnew);
        float p0 = __expf(a0 - mnew), p1 = __expf(a1 - mnew);
        float rs = p0 + p1;
        for (int off = 1; off < 16; off <<= 1) rs += __shfl_xor(rs, off, 32);
        lrun[r] = lrun[r] * corr + rs;
        mrun[r] = mnew;
        O0[r] *= corr; O1[r] *= corr;
        int m = r + roff;
        P[m * 32 + col] = (_Float16)p0;
        P[m * 32 + 16 + col] = (_Float16)p1;
      }
      v16h pf = load_a_frag(&Pbuf[wave][0], 32, 0, 0, lane);
      O0 = wmma_f16(pf, load_b_frag(Vsw, 512, 0, key0, lane), O0);
      O1 = wmma_f16(pf, load_b_frag(Vsw, 512, 16, key0, lane), O1);
    }
#pragma unroll
    for (int r = 0; r < 8; ++r) {
      float inv = 1.0f / lrun[r];
      size_t s = (size_t)b * 512 + q0 + r + roff;
      attnh[s * 128 + h * 32 + col] = (_Float16)(O0[r] * inv);
      attnh[s * 128 + h * 32 + 16 + col] = (_Float16)(O1[r] * inv);
    }
  }
}

// ===========================================================================
// 7) Mean pool over S, then the small fused head.
// ===========================================================================
__global__ void pool_kernel(const float* __restrict__ attnout,
                            float* __restrict__ pooled) {
  int b = blockIdx.x, e = threadIdx.x;
  float s = 0.0f;
  for (int t = 0; t < 512; ++t) s += attnout[((size_t)b * 512 + t) * 128 + e];
  pooled[b * 128 + e] = s * (1.0f / 512.0f);
}

__global__ void head_kernel(const float* __restrict__ pooled,
                            const float* __restrict__ es,
                            const float* gpW, const float* gpb,
                            const float* espW, const float* espb,
                            const float* fusW1, const float* fusb1,
                            const float* fusW2, const float* fusb2,
                            const float* outW, const float* outb,
                            const float* combw, float* __restrict__ out) {
  __shared__ float comb[67];
  __shared__ float fused1[128];
  __shared__ float fused2[64];
  int b = blockIdx.x, tid = threadIdx.x;
  const float* pb = pooled + b * 128;
  if (tid < 64) {
    float a = gpb[tid];
    for (int k = 0; k < 128; ++k) a += pb[k] * gpW[tid * 128 + k];
    comb[tid] = fmaxf(a, 0.0f);
  }
  if (tid < 3) comb[64 + tid] = es[((size_t)b * 512 + 511) * 3 + tid];
  __syncthreads();
  {
    float a = fusb1[tid];
    for (int k = 0; k < 67; ++k) a += comb[k] * fusW1[tid * 67 + k];
    fused1[tid] = fmaxf(a, 0.0f);
  }
  __syncthreads();
  if (tid < 64) {
    float a = fusb2[tid];
    for (int k = 0; k < 128; ++k) a += fused1[k] * fusW2[tid * 128 + k];
    fused2[tid] = fmaxf(a, 0.0f);
  }
  __syncthreads();
  if (tid < 5) {
    float r = outb[tid];
    for (int k = 0; k < 64; ++k) r += fused2[k] * outW[tid * 64 + k];
    float e0 = es[((size_t)b * 512 + 511) * 3 + 0];
    float e1 = es[((size_t)b * 512 + 511) * 3 + 1];
    float e2 = es[((size_t)b * 512 + 511) * 3 + 2];
    float ef = tanh_fast(e0 * espW[tid * 3 + 0] + e1 * espW[tid * 3 + 1] +
                         e2 * espW[tid * 3 + 2] + espb[tid]);
    float w = sigf(combw[0]);
    out[b * 5 + tid] = w * ef + (1.0f - w) * r;
  }
}

// ===========================================================================
// Launch
// ===========================================================================
extern "C" void kernel_launch(void* const* d_in, const int* in_sizes, int n_in,
                              void* d_out, int out_size, void* d_ws,
                              size_t ws_size, hipStream_t stream) {
  const float* x      = (const float*)d_in[0];
  const float* alpha  = (const float*)d_in[1];
  const float* beta   = (const float*)d_in[2];
  const float* gamma  = (const float*)d_in[3];
  const float* projW  = (const float*)d_in[4];
  const float* projb  = (const float*)d_in[5];
  const float* Wih0   = (const float*)d_in[6];
  const float* Whh0   = (const float*)d_in[7];
  const float* bih0   = (const float*)d_in[8];
  const float* bhh0   = (const float*)d_in[9];
  const float* Wih1   = (const float*)d_in[10];
  const float* Whh1   = (const float*)d_in[11];
  const float* bih1   = (const float*)d_in[12];
  const float* bhh1   = (const float*)d_in[13];
  const float* Wqkv   = (const float*)d_in[14];
  const float* bqkv   = (const float*)d_in[15];
  const float* Wo     = (const float*)d_in[16];
  const float* bo     = (const float*)d_in[17];
  const float* gpW    = (const float*)d_in[18];
  const float* gpb    = (const float*)d_in[19];
  const float* fusW1  = (const float*)d_in[20];
  const float* fusb1  = (const float*)d_in[21];
  const float* fusW2  = (const float*)d_in[22];
  const float* fusb2  = (const float*)d_in[23];
  const float* outW   = (const float*)d_in[24];
  const float* outb   = (const float*)d_in[25];
  const float* espW   = (const float*)d_in[26];
  const float* espb   = (const float*)d_in[27];
  const float* combw  = (const float*)d_in[28];
  float* out = (float*)d_out;

  // --- workspace carve (all offsets 256B aligned) ---
  char* w = (char*)d_ws;
  size_t off = 0;
  auto carve = [&](size_t bytes) {
    void* p = w + off;
    off += (bytes + 255) & ~(size_t)255;
    return p;
  };
  float*     es      = (float*)carve((size_t)64 * 512 * 3 * 4);
  float*     weff    = (float*)carve(512 * 4 * 4);
  float*     bias0   = (float*)carve(512 * 4);
  float*     bias1   = (float*)carve(512 * 4);
  _Float16*  whh0h   = (_Float16*)carve(512 * 128 * 2);
  _Float16*  whh1h   = (_Float16*)carve(512 * 128 * 2);
  _Float16*  wih1h   = (_Float16*)carve(512 * 128 * 2);
  _Float16*  wqkvh   = (_Float16*)carve(384 * 128 * 2);
  _Float16*  woh     = (_Float16*)carve(128 * 128 * 2);
  float*     xg      = (float*)carve((size_t)64 * 512 * 512 * 4);  // [t][b][n]
  _Float16*  h1h     = (_Float16*)carve((size_t)64 * 512 * 128 * 2);
  _Float16*  h2h     = (_Float16*)carve((size_t)64 * 512 * 128 * 2);
  _Float16*  qkvh    = (_Float16*)carve((size_t)64 * 512 * 384 * 2);
  _Float16*  attnhb  = (_Float16*)carve((size_t)64 * 512 * 128 * 2);
  float*     attnout = (float*)carve((size_t)64 * 512 * 128 * 4);
  float*     pooled  = (float*)carve(64 * 128 * 4);
  (void)ws_size; (void)n_in; (void)in_sizes; (void)out_size;

  auto gemm_blocks = [](int M, int N) {
    int tiles = (M / 16) * (N / 16);
    return (tiles + 7) / 8;  // 8 waves (256 threads) per block
  };
  const int lstm_lds = (64 * 128 + 512 * 128) * 2;  // 144KB dynamic LDS

  // 1) ES scan
  es_kernel<<<1, 64, 0, stream>>>(x, alpha, beta, gamma, es);
  // 2) prep: folded weights + combined biases + f16 weight conversions
  weff_kernel<<<2, 256, 0, stream>>>(Wih0, projW, projb, bih0, bhh0, bih1, bhh1,
                                     weff, bias0, bias1);
  f32_to_f16<<<(512 * 128 + 255) / 256, 256, 0, stream>>>(Whh0, whh0h, 512 * 128);
  f32_to_f16<<<(512 * 128 + 255) / 256, 256, 0, stream>>>(Whh1, whh1h, 512 * 128);
  f32_to_f16<<<(512 * 128 + 255) / 256, 256, 0, stream>>>(Wih1, wih1h, 512 * 128);
  f32_to_f16<<<(384 * 128 + 255) / 256, 256, 0, stream>>>(Wqkv, wqkvh, 384 * 128);
  f32_to_f16<<<(128 * 128 + 255) / 256, 256, 0, stream>>>(Wo, woh, 128 * 128);
  // 3) gate inputs, layer 0 (folded K=7 contraction), time-major out
  xg0_kernel<<<(int)(((size_t)32768 * 512 + 255) / 256), 256, 0, stream>>>(
      x, es, weff, Wih0, bias0, xg);
  // 4) LSTM layer 0 (persistent WMMA recurrence, Whh in LDS via TDM)
  lstm_kernel<<<1, 512, lstm_lds, stream>>>(xg, whh0h, h1h);
  // 5) gate inputs, layer 1: xg = h1 @ Wih1.T + bias1 (WMMA GEMM, time-major)
  gemm_kernel<false, true, 128><<<gemm_blocks(32768, 512), 256, 0, stream>>>(
      h1h, wih1h, bias1, xg, 32768, 512);
  // 6) LSTM layer 1
  lstm_kernel<<<1, 512, lstm_lds, stream>>>(xg, whh1h, h2h);
  // 7) QKV projection (WMMA GEMM, f16 out)
  gemm_kernel<true, false, 128><<<gemm_blocks(32768, 384), 256, 0, stream>>>(
      h2h, wqkvh, bqkv, qkvh, 32768, 384);
  // 8) flash attention per (b, head)
  attn_kernel<<<256, 256, 0, stream>>>(qkvh, attnhb);
  // 9) output projection (WMMA GEMM, f32 out)
  gemm_kernel<false, false, 128><<<gemm_blocks(32768, 128), 256, 0, stream>>>(
      attnhb, woh, bo, attnout, 32768, 128);
  // 10) mean pool + head
  pool_kernel<<<64, 128, 0, stream>>>(attnout, pooled);
  head_kernel<<<64, 128, 0, stream>>>(pooled, es, gpW, gpb, espW, espb, fusW1,
                                      fusb1, fusW2, fusb2, outW, outb, combw,
                                      out);
}
